// Layer_28793460752999
// MI455X (gfx1250) — matrix operand order repaired and checked by reading, not compiled, not run on hardware
//
#include <hip/hip_runtime.h>

// ---------------------------------------------------------------------------
// Shapes: B=16, CUTS=8, CUT=6, N=1024, D=64, K=8 heads, HS=8, MEM=10
// ---------------------------------------------------------------------------
#define BB   16
#define CUTS 8
#define NN   1024
#define DD   64
#define TT   7            // 1 (emb) + 6 (cut)
#define ROWS (BB * NN)    // 16384 rows for gating / projections
#define HID  4096         // D*D hidden dim of the gate MLPs
#define RSTRIDE (NN * DD) // 65536: row stride (b*7+t) in tbuf/kbuf/vbuf
#define FRAG    512       // 32 lanes * 16 bf16 elems per packed B fragment

typedef __attribute__((ext_vector_type(16))) __bf16 v16bf;
typedef __attribute__((ext_vector_type(8)))  __bf16 v8bf;
typedef __attribute__((ext_vector_type(8)))  float  v8f;
typedef __attribute__((ext_vector_type(4)))  float  v4f;

__device__ __forceinline__ v8f wmma_bf16(v16bf a, v16bf b, v8f c) {
  return __builtin_amdgcn_wmma_f32_16x16x32_bf16(
      /*neg_a=*/false, a, /*neg_b=*/false, b,
      /*c_mod=*/(short)0, c, /*reuse_a=*/false, /*reuse_b=*/false);
}

__device__ __forceinline__ float fast_tanh(float x) {
#if __has_builtin(__builtin_amdgcn_tanhf)
  return __builtin_amdgcn_tanhf(x);   // v_tanh_f32 (CDNA5 TRANS op)
#else
  return tanhf(x);
#endif
}
__device__ __forceinline__ float fast_sigmoid(float x) {
  return 1.f / (1.f + __expf(-x));
}

// A fragment (16x32, bf16) from f32 row-major source; 16B-aligned vector loads.
// Layout: lanes 0-15 -> rows, K = {0..7,16..23}; lanes 16-31 -> K = {8..15,24..31}.
__device__ __forceinline__ v16bf load_a_f32(const float* __restrict__ src,
                                            long ldm, int k0, int lane) {
  int row = lane & 15;
  int kb  = (lane < 16) ? 0 : 8;
  const v4f* q = (const v4f*)(src + (long)row * ldm + k0 + kb);
  v4f f0 = q[0], f1 = q[1], f2 = q[4], f3 = q[5];
  v16bf a;
#pragma unroll
  for (int e = 0; e < 4; ++e) {
    a[e]      = (__bf16)f0[e];
    a[4 + e]  = (__bf16)f1[e];
    a[8 + e]  = (__bf16)f2[e];
    a[12 + e] = (__bf16)f3[e];
  }
  return a;
}

// A fragment from bf16 row-major LDS (two aligned 16B loads).
__device__ __forceinline__ v16bf load_a_lds(const __bf16* __restrict__ src,
                                            int ldm, int k0, int lane) {
  int row = lane & 15;
  int kb  = (lane < 16) ? 0 : 8;
  const __bf16* p = src + row * ldm + k0 + kb;
  v8bf lo = *(const v8bf*)(p);
  v8bf hi = *(const v8bf*)(p + 16);
  return __builtin_shufflevector(lo, hi, 0, 1, 2, 3, 4, 5, 6, 7,
                                 8, 9, 10, 11, 12, 13, 14, 15);
}

// Packed B fragment: one contiguous 32-byte load per lane.
__device__ __forceinline__ v16bf load_b_packed(const __bf16* __restrict__ frag,
                                               int lane) {
  return *(const v16bf*)(frag + lane * 16);
}

// ---------------------------------------------------------------------------
// Pack a row-major f32 (K x N) weight matrix into WMMA-B fragment layout:
// frag_id = nt*(K/32) + kt ; within frag: [lane][elem] contiguous (bf16).
// lanes 0-15: col, K=kt*32+{0..7,16..23} ; lanes 16-31: col, K=kt*32+{8..15,24..31}.
// ---------------------------------------------------------------------------
__global__ __launch_bounds__(256) void pack_b_kernel(
    const float* __restrict__ src, __bf16* __restrict__ dst, int K, int N) {
  int j = blockIdx.x * 256 + threadIdx.x;
  if (j >= K * N) return;
  int k = j / N, nn = j % N;
  int kt = k >> 5, kk = k & 31;
  int lane = (nn & 15) + ((kk & 8) ? 16 : 0);
  int elem = (kk & 7) + ((kk & 16) ? 8 : 0);
  int nt = nn >> 4;
  int KT = K >> 5;
  long idx = (((long)nt * KT + kt) * 32 + lane) * 16 + elem;
  dst[idx] = (__bf16)src[j];
}

// ---------------------------------------------------------------------------
// mem = tanh(tanh(memory@Wg1+bg1)@Wg2+bg2)@Wg3+bg3      (1024 x 100)
// ---------------------------------------------------------------------------
__global__ __launch_bounds__(128) void mem_kernel(
    const float* __restrict__ memory, const float* __restrict__ Wg1,
    const float* __restrict__ bg1, const float* __restrict__ Wg2,
    const float* __restrict__ bg2, const float* __restrict__ Wg3,
    const float* __restrict__ bg3, float* __restrict__ mem_out) {
  __shared__ float mrow[5];
  __shared__ float h1[64];
  __shared__ float h2[64];
  int n = blockIdx.x, tid = threadIdx.x;
  if (tid < 5) mrow[tid] = memory[n * 5 + tid];
  __syncthreads();
  if (tid < 64) {
    float s = bg1[tid];
#pragma unroll
    for (int c = 0; c < 5; ++c) s += mrow[c] * Wg1[c * 64 + tid];
    h1[tid] = fast_tanh(s);
  }
  __syncthreads();
  if (tid < 64) {
    float s = bg2[tid];
    for (int c = 0; c < 64; ++c) s += h1[c] * Wg2[c * 64 + tid];
    h2[tid] = fast_tanh(s);
  }
  __syncthreads();
  for (int j = tid; j < 100; j += 128) {
    float s = bg3[j];
    for (int c = 0; c < 64; ++c) s += h2[c] * Wg3[c * 100 + j];
    mem_out[n * 100 + j] = s;
  }
}

// ---------------------------------------------------------------------------
// Wd[k][n] = P_d[k](64x10) @ memV[n](10x10) @ Q_d[k](10x64)  -> packed bf16
// bd[k][n][e] = sum_m mem[n][m] * Bm_d[k][m][e]              -> f32
// ---------------------------------------------------------------------------
__global__ __launch_bounds__(128) void wd_kernel(
    const float* __restrict__ mem, const float* __restrict__ P_d,
    const float* __restrict__ Q_d, const float* __restrict__ Bm_d,
    __bf16* __restrict__ wd_pk, float* __restrict__ bd) {
  __shared__ float mv[100];
  __shared__ float tmp[640];  // (64 x 10) = P_d[k] @ memV
  int n = blockIdx.x, tid = threadIdx.x;
  for (int j = tid; j < 100; j += 128) mv[j] = mem[n * 100 + j];
  __syncthreads();
  for (int k = 0; k < 2; ++k) {
    for (int j = tid; j < 640; j += 128) {
      int d = j / 10, p = j % 10;
      float s = 0.f;
#pragma unroll
      for (int m = 0; m < 10; ++m) s += P_d[(k * 64 + d) * 10 + m] * mv[m * 10 + p];
      tmp[j] = s;
    }
    __syncthreads();
    for (int j = tid; j < 4096; j += 128) {
      int d = j >> 6, e = j & 63;  // d = K index, e = N index
      float s = 0.f;
#pragma unroll
      for (int p = 0; p < 10; ++p) s += tmp[d * 10 + p] * Q_d[(k * 10 + p) * 64 + e];
      // pack into B-fragment layout (KT = 2, NT = 4)
      int kt = d >> 5, kk = d & 31;
      int lane = (e & 15) + ((kk & 8) ? 16 : 0);
      int elem = (kk & 7) + ((kk & 16) ? 8 : 0);
      int nt = e >> 4;
      long idx = ((long)(nt * 2 + kt) * 32 + lane) * 16 + elem;
      wd_pk[(long)(k * NN + n) * 4096 + idx] = (__bf16)s;
    }
    for (int j = tid; j < 64; j += 128) {
      float s = 0.f;
      for (int m = 0; m < 100; ++m) s += mv[m] * Bm_d[(k * 100 + m) * 64 + j];
      bd[(k * NN + n) * 64 + j] = s;
    }
    __syncthreads();
  }
}

// ---------------------------------------------------------------------------
// Gate: out_new = tanh(tanh(o@W1a+b1a)@W1b+b1b) * sigmoid(tanh(o@W2a+b2a)@W2b+b2b)
// 32 rows (2 M-tiles) per block; 4 waves; hidden staged through 128KB LDS
// (bf16) in two 2048-column halves; every B fragment reused for 2 M-tiles.
// ---------------------------------------------------------------------------
__global__ __launch_bounds__(128) void gate_kernel(
    const float* __restrict__ outbuf,
    const __bf16* __restrict__ w1a, const float* __restrict__ b1a,
    const __bf16* __restrict__ w1b, const float* __restrict__ b1b,
    const __bf16* __restrict__ w2a, const float* __restrict__ b2a,
    const __bf16* __restrict__ w2b, const float* __restrict__ b2b,
    float* __restrict__ gated) {
  __shared__ __align__(32) __bf16 hlds[32 * 2048];  // 128 KB
  const int tile = blockIdx.x;        // 0..511: rows [tile*32, tile*32+32)
  const int wave = threadIdx.x >> 5;  // 0..3
  const int lane = threadIdx.x & 31;
  const int col = lane & 15;
  const int row_off = (lane < 16) ? 0 : 8;

  const float* arows = outbuf + (long)tile * 32 * DD;
  v16bf a0[2], a1[2];
#pragma unroll
  for (int m = 0; m < 2; ++m) {
    a0[m] = load_a_f32(arows + (long)m * 16 * DD, DD, 0, lane);
    a1[m] = load_a_f32(arows + (long)m * 16 * DD, DD, 32, lane);
  }

  const int n0o = wave * 16;  // this wave's output-tile columns
  v8f g1v[2], g2v[2];

  for (int path = 0; path < 2; ++path) {
    const __bf16* wa = path ? w2a : w1a;
    const float*  ba = path ? b2a : b1a;
    const __bf16* wb = path ? w2b : w1b;
    const float*  bb = path ? b2b : b1b;

    v8f c2[2] = {v8f{}, v8f{}};
    for (int half = 0; half < 2; ++half) {
      const int base = half * 2048;
      // --- GEMM1: hidden[32 x 2048] = tanh(A(32x64) @ Wa + ba) ---
      for (int j = 0; j < 32; ++j) {
        int nt = half * 128 + wave * 32 + j;   // hidden column tile (KT=2)
        const __bf16* fb = wa + (long)nt * 2 * FRAG;
        v16bf b0 = load_b_packed(fb, lane);
        v16bf b1 = load_b_packed(fb + FRAG, lane);
        int n0 = nt * 16;
        float bias = ba[n0 + col];
        int lc = n0 - base + col;
#pragma unroll
        for (int m = 0; m < 2; ++m) {
          v8f c = {};
          c = wmma_bf16(a0[m], b0, c);
          c = wmma_bf16(a1[m], b1, c);
#pragma unroll
          for (int v = 0; v < 8; ++v)
            hlds[(m * 16 + v + row_off) * 2048 + lc] =
                (__bf16)fast_tanh(c[v] + bias);
        }
      }
      __syncthreads();
      // --- GEMM2: accumulate G(32x64) over this half's K = 2048 ---
      // Wb packed frags: frag_id = nt*128 + kt (KT = 128, NT = 4)
      const __bf16* fbb = wb + ((long)wave * 128 + half * 64) * FRAG;
#pragma unroll 2
      for (int ks = 0; ks < 64; ++ks) {
        v16bf bh = load_b_packed(fbb + (long)ks * FRAG, lane);
#pragma unroll
        for (int m = 0; m < 2; ++m) {
          v16bf ah = load_a_lds(hlds + m * 16 * 2048, 2048, ks * 32, lane);
          c2[m] = wmma_bf16(ah, bh, c2[m]);
        }
      }
      __syncthreads();
    }
    float bias = bb[n0o + col];
#pragma unroll
    for (int m = 0; m < 2; ++m) {
      if (path == 0) {
#pragma unroll
        for (int v = 0; v < 8; ++v) g1v[m][v] = fast_tanh(c2[m][v] + bias);
      } else {
#pragma unroll
        for (int v = 0; v < 8; ++v) g2v[m][v] = fast_sigmoid(c2[m][v] + bias);
      }
    }
  }
#pragma unroll
  for (int m = 0; m < 2; ++m)
#pragma unroll
    for (int v = 0; v < 8; ++v)
      gated[(long)(tile * 32 + m * 16 + v + row_off) * DD + n0o + col] =
          g1v[m][v] * g2v[m][v];
}

// ---------------------------------------------------------------------------
// Build t = concat(emb (+gated), x[:, cut*6:(cut+1)*6]) -> tbuf[b][t][n][d]
// ---------------------------------------------------------------------------
__global__ __launch_bounds__(256) void prep_t_kernel(
    const float* __restrict__ x, const float* __restrict__ te,
    const float* __restrict__ W_emb, const float* __restrict__ b_emb,
    const float* __restrict__ gated, int cut, float* __restrict__ tbuf) {
  long idx = (long)blockIdx.x * 256 + threadIdx.x;  // flat index into tbuf
  int d = idx & 63;
  long r = idx >> 6;
  int n = r & 1023;
  long bt = r >> 10;       // b*7 + t
  int t = (int)(bt % 7);
  int b = (int)(bt / 7);
  float val;
  if (t == 0) {
    float s = b_emb[cut * 64 + d];
#pragma unroll
    for (int c = 0; c < 5; ++c)
      s += te[(cut * NN + n) * 5 + c] * W_emb[(cut * 5 + c) * 64 + d];
    if (gated) s += gated[(long)(b * NN + n) * 64 + d];
    val = s;
  } else {
    val = x[(((long)b * 48 + cut * 6 + (t - 1)) * NN + n) * 64 + d];
  }
  tbuf[idx] = val;
}

// ---------------------------------------------------------------------------
// Per-position K/V transform: k = t @ Wd[0][n] + bd[0][n], v = t @ Wd[1][n] + bd[1][n]
// Each wave holds A fragments for up to 2 M-tiles in registers and reuses
// every B fragment across both.
// ---------------------------------------------------------------------------
__global__ __launch_bounds__(128) void kv_kernel(
    const float* __restrict__ tbuf, const __bf16* __restrict__ wd_pk,
    const float* __restrict__ bd, float* __restrict__ kbuf,
    float* __restrict__ vbuf) {
  const int n = blockIdx.x;
  const int wave = threadIdx.x >> 5;
  const int lane = threadIdx.x & 31;
  const int col = lane & 15;
  const int row_off = (lane < 16) ? 0 : 8;

  const int nmt = (wave + 4 < 7) ? 2 : 1;  // M-tiles: {wave, wave+4} over 112 rows
  v16bf a0[2], a1[2];
  for (int m = 0; m < nmt; ++m) {
    const float* asrc =
        tbuf + (long)((wave + 4 * m) * 16) * RSTRIDE + (long)n * DD;
    a0[m] = load_a_f32(asrc, RSTRIDE, 0, lane);
    a1[m] = load_a_f32(asrc, RSTRIDE, 32, lane);
  }
#pragma unroll
  for (int mat = 0; mat < 2; ++mat) {
    const __bf16* base = wd_pk + (long)(mat * NN + n) * 4096;
    float* dst = mat ? vbuf : kbuf;
#pragma unroll
    for (int nt = 0; nt < 4; ++nt) {
      v16bf b0 = load_b_packed(base + (nt * 2 + 0) * FRAG, lane);
      v16bf b1 = load_b_packed(base + (nt * 2 + 1) * FRAG, lane);
      float bias = bd[(mat * NN + n) * 64 + nt * 16 + col];
      for (int m = 0; m < nmt; ++m) {
        v8f c = {};
        c = wmma_bf16(a0[m], b0, c);
        c = wmma_bf16(a1[m], b1, c);
#pragma unroll
        for (int v = 0; v < 8; ++v) {
          int rr = (wave + 4 * m) * 16 + v + row_off;
          dst[(long)rr * RSTRIDE + (long)n * DD + nt * 16 + col] = c[v] + bias;
        }
      }
    }
  }
}

// ---------------------------------------------------------------------------
// Attention (T=7, 8 heads of 8) + tanh(x@P0+B0)@P1+B1. One (b,n) per block.
// ---------------------------------------------------------------------------
__global__ __launch_bounds__(64) void attn_kernel(
    const float* __restrict__ tbuf, const float* __restrict__ kbuf,
    const float* __restrict__ vbuf, const float* __restrict__ P_sh,
    const float* __restrict__ B_sh, float* __restrict__ outbuf,
    float* __restrict__ dout, int cut) {
  __shared__ float qrow[64];
  __shared__ float pr[8][7];
  __shared__ float xrow[64];
  __shared__ float y1[64];
  const int bn = blockIdx.x;
  const int b = bn >> 10, n = bn & 1023;
  const int tid = threadIdx.x;
  const long base = (long)(b * 7) * RSTRIDE + (long)n * DD;

  qrow[tid] = tbuf[base + tid];  // q = t[:, 0]
  __syncthreads();
  if (tid < 56) {
    int h = tid / 7, t = tid % 7;
    const float* kp = kbuf + base + (long)t * RSTRIDE + h * 8;
    float s = 0.f;
#pragma unroll
    for (int e = 0; e < 8; ++e) s += qrow[h * 8 + e] * kp[e];
    pr[h][t] = s * 0.35355339059327373f;  // 1/sqrt(8)
  }
  __syncthreads();
  if (tid < 8) {
    float mx = -1e30f;
#pragma unroll
    for (int t = 0; t < 7; ++t) mx = fmaxf(mx, pr[tid][t]);
    float sum = 0.f;
#pragma unroll
    for (int t = 0; t < 7; ++t) {
      float e = __expf(pr[tid][t] - mx);
      pr[tid][t] = e;
      sum += e;
    }
    float inv = 1.f / sum;
#pragma unroll
    for (int t = 0; t < 7; ++t) pr[tid][t] *= inv;
  }
  __syncthreads();
  {
    int h = tid >> 3;
    float s = 0.f;
#pragma unroll
    for (int t = 0; t < 7; ++t) s += pr[h][t] * vbuf[base + (long)t * RSTRIDE + tid];
    xrow[tid] = s;
  }
  __syncthreads();
  {
    float s = B_sh[tid];
    for (int d = 0; d < 64; ++d) s += xrow[d] * P_sh[d * 64 + tid];
    y1[tid] = fast_tanh(s);
  }
  __syncthreads();
  {
    float s = B_sh[64 + tid];
    for (int d = 0; d < 64; ++d) s += y1[d] * P_sh[4096 + d * 64 + tid];
    outbuf[(long)(b * NN + n) * 64 + tid] = s;
    dout[(long)((b * CUTS + cut) * NN + n) * 64 + tid] = s;
  }
}

// ---------------------------------------------------------------------------
// Host orchestration
// ---------------------------------------------------------------------------
static inline char* ws_alloc(void* ws, size_t& off, size_t bytes) {
  char* p = (char*)ws + off;
  off += (bytes + 255) & ~(size_t)255;
  return p;
}

extern "C" void kernel_launch(void* const* d_in, const int* in_sizes, int n_in,
                              void* d_out, int out_size, void* d_ws,
                              size_t ws_size, hipStream_t stream) {
  const float* x      = (const float*)d_in[0];
  const float* te     = (const float*)d_in[1];
  const float* W_emb  = (const float*)d_in[2];
  const float* b_emb  = (const float*)d_in[3];
  const float* W1a    = (const float*)d_in[4];
  const float* b1a    = (const float*)d_in[5];
  const float* W1b    = (const float*)d_in[6];
  const float* b1b    = (const float*)d_in[7];
  const float* W2a    = (const float*)d_in[8];
  const float* b2a    = (const float*)d_in[9];
  const float* W2b    = (const float*)d_in[10];
  const float* b2b    = (const float*)d_in[11];
  const float* P_sh   = (const float*)d_in[12];
  const float* B_sh   = (const float*)d_in[13];
  const float* memory = (const float*)d_in[14];
  const float* Wg1    = (const float*)d_in[15];
  const float* bg1    = (const float*)d_in[16];
  const float* Wg2    = (const float*)d_in[17];
  const float* bg2    = (const float*)d_in[18];
  const float* Wg3    = (const float*)d_in[19];
  const float* bg3    = (const float*)d_in[20];
  const float* P_d    = (const float*)d_in[21];
  const float* Q_d    = (const float*)d_in[22];
  const float* Bm_d   = (const float*)d_in[23];
  float* out = (float*)d_out;

  size_t off = 0;
  float*  mem_f  = (float*)ws_alloc(d_ws, off, (size_t)NN * 100 * 4);
  float*  bd_f   = (float*)ws_alloc(d_ws, off, (size_t)2 * NN * 64 * 4);
  __bf16* wd_pk  = (__bf16*)ws_alloc(d_ws, off, (size_t)2 * NN * 4096 * 2);
  __bf16* w1a_pk = (__bf16*)ws_alloc(d_ws, off, (size_t)DD * HID * 2);
  __bf16* w1b_pk = (__bf16*)ws_alloc(d_ws, off, (size_t)HID * DD * 2);
  __bf16* w2a_pk = (__bf16*)ws_alloc(d_ws, off, (size_t)DD * HID * 2);
  __bf16* w2b_pk = (__bf16*)ws_alloc(d_ws, off, (size_t)HID * DD * 2);
  float*  tbuf   = (float*)ws_alloc(d_ws, off, (size_t)BB * TT * NN * DD * 4);
  float*  kbuf   = (float*)ws_alloc(d_ws, off, (size_t)BB * TT * NN * DD * 4);
  float*  vbuf   = (float*)ws_alloc(d_ws, off, (size_t)BB * TT * NN * DD * 4);
  float*  outbuf = (float*)ws_alloc(d_ws, off, (size_t)ROWS * DD * 4);
  float*  gated  = (float*)ws_alloc(d_ws, off, (size_t)ROWS * DD * 4);
  (void)ws_size;

  // Precompute: mem MLP, per-position Wd/bd, packed bf16 weight fragments.
  mem_kernel<<<NN, 128, 0, stream>>>(memory, Wg1, bg1, Wg2, bg2, Wg3, bg3, mem_f);
  wd_kernel<<<NN, 128, 0, stream>>>(mem_f, P_d, Q_d, Bm_d, wd_pk, bd_f);
  pack_b_kernel<<<(DD * HID + 255) / 256, 256, 0, stream>>>(W1a, w1a_pk, DD, HID);
  pack_b_kernel<<<(HID * DD + 255) / 256, 256, 0, stream>>>(W1b, w1b_pk, HID, DD);
  pack_b_kernel<<<(DD * HID + 255) / 256, 256, 0, stream>>>(W2a, w2a_pk, DD, HID);
  pack_b_kernel<<<(HID * DD + 255) / 256, 256, 0, stream>>>(W2b, w2b_pk, HID, DD);

  const int prep_blocks = (BB * TT * NN * DD) / 256;
  for (int i = 0; i < CUTS; ++i) {
    if (i > 0) {
      gate_kernel<<<ROWS / 32, 128, 0, stream>>>(
          outbuf, w1a_pk, b1a, w1b_pk, b1b, w2a_pk, b2a, w2b_pk, b2b, gated);
    }
    prep_t_kernel<<<prep_blocks, 256, 0, stream>>>(
        x, te, W_emb, b_emb, (i > 0) ? gated : (const float*)nullptr, i, tbuf);
    kv_kernel<<<NN, 128, 0, stream>>>(tbuf, wd_pk, bd_f, kbuf, vbuf);
    attn_kernel<<<BB * NN, 64, 0, stream>>>(tbuf, kbuf, vbuf, P_sh, B_sh,
                                            outbuf, out, i);
  }
  (void)in_sizes; (void)n_in; (void)out_size;
}
